// MS_Loss_80444737454309
// MI455X (gfx1250) — matrix operand order repaired and checked by reading, not compiled, not run on hardware
//
#include <hip/hip_runtime.h>
#include <hip/hip_bf16.h>
#include <stdint.h>

#define D_SZ 1024
#define TWO_D 2048
#define IN_D 64
#define NHEAD 4
#define NBATCH 8

typedef __bf16 v16bf __attribute__((ext_vector_type(16)));
typedef __bf16 v8bf  __attribute__((ext_vector_type(8)));
typedef float  v8f   __attribute__((ext_vector_type(8)));
// Matches the builtin's parameter pointee type exactly (gcc vector_size style).
typedef int v4i_vs __attribute__((vector_size(16)));

#if defined(__gfx1250__) && __has_builtin(__builtin_amdgcn_global_load_async_to_lds_b128)
#define USE_ASYNC 1
#else
#define USE_ASYNC 0
#endif

// Copy 16B from per-lane global address to per-lane LDS address.
// Async path uses GLOBAL_LOAD_ASYNC_TO_LDS_B128 (tracked by ASYNCcnt).
__device__ __forceinline__ void copy_b128(const void* g, void* l) {
#if USE_ASYNC
  auto gp = (__attribute__((address_space(1))) v4i_vs*)(unsigned long long)(uintptr_t)g;
  auto lp = (__attribute__((address_space(3))) v4i_vs*)(unsigned int)(uintptr_t)l;
  __builtin_amdgcn_global_load_async_to_lds_b128(gp, lp, 0, 0);
#else
  *(uint4*)l = *(const uint4*)g;
#endif
}

__device__ __forceinline__ void wait_async_copies() {
#if USE_ASYNC
#if __has_builtin(__builtin_amdgcn_s_wait_asynccnt)
  __builtin_amdgcn_s_wait_asynccnt(0);
#else
  asm volatile("s_wait_asynccnt 0x0" ::: "memory");
#endif
#endif
}

__device__ __forceinline__ __bf16 f2bf(float x) {
  uint32_t u = __builtin_bit_cast(uint32_t, x);
  uint32_t r = u + 0x7FFFu + ((u >> 16) & 1u);
  uint16_t h = (uint16_t)(r >> 16);
  return __builtin_bit_cast(__bf16, h);
}
__device__ __forceinline__ float bf2f(__bf16 b) {
  uint32_t u = ((uint32_t)__builtin_bit_cast(uint16_t, b)) << 16;
  return __builtin_bit_cast(float, u);
}

// A-matrix 16x32 bf16 fragment from row-major LDS tile (row stride lda elems).
// Lane l: M = l%16, half = l/16. VGPR0-3: K = half*8 + 0..7 ; VGPR4-7: K = 16 + half*8 + 0..7.
__device__ __forceinline__ v16bf load_frag_a(const __bf16* p, int lda) {
  int lane = threadIdx.x & 31;
  const __bf16* base = p + (lane & 15) * lda + ((lane >> 4) << 3);
  v8bf lo = *(const v8bf*)base;
  v8bf hi = *(const v8bf*)(base + 16);
  v16bf f;
#pragma unroll
  for (int e = 0; e < 8; e++) { f[e] = lo[e]; f[8 + e] = hi[e]; }
  return f;
}

// B-matrix 32x16 bf16 fragment from N-major LDS tile (lB[n][k], row stride lda).
// Lane l: N = l%16, half = l/16. VGPR0-3: K = half*8 + 0..7 ; VGPR4-7: K = 16 + half*8 + 0..7.
__device__ __forceinline__ v16bf load_frag_b(const __bf16* p, int lda) {
  int lane = threadIdx.x & 31;
  const __bf16* base = p + (lane & 15) * lda + ((lane >> 4) << 3);
  v8bf lo = *(const v8bf*)base;
  v8bf hi = *(const v8bf*)(base + 16);
  v16bf f;
#pragma unroll
  for (int e = 0; e < 8; e++) { f[e] = lo[e]; f[8 + e] = hi[e]; }
  return f;
}

__device__ __forceinline__ v8f wmma_bf16(v16bf a, v16bf b, v8f c) {
  return __builtin_amdgcn_wmma_f32_16x16x32_bf16(false, a, false, b, (short)0, c, false, false);
}

// ---------------- Kernel 1: edge MLPs -> weight^T, weightf^T (bf16) ----------------
__global__ void __launch_bounds__(256) mlp_edge_kernel(
    const float* kW1, const float* kb1, const float* kW2, const float* kb2,
    const float* kW3, const float* kb3,
    const float* fW1, const float* fb1, const float* fW2, const float* fb2,
    const float* fW3, const float* fb3,
    __bf16* wT, __bf16* wfT)
{
  __shared__ float sW1[2][4][32];
  __shared__ float sB1[2][32];
  __shared__ float sW2[2][32][64];
  __shared__ float sB2[2][64];
  __shared__ float sW3[2][64];
  __shared__ float sB3[2];
  int tid = threadIdx.x;
  for (int i = tid; i < 128; i += 256) { sW1[0][i >> 5][i & 31] = kW1[i]; sW1[1][i >> 5][i & 31] = fW1[i]; }
  for (int i = tid; i < 32; i += 256)  { sB1[0][i] = kb1[i]; sB1[1][i] = fb1[i]; }
  for (int i = tid; i < 2048; i += 256){ sW2[0][i >> 6][i & 63] = kW2[i]; sW2[1][i >> 6][i & 63] = fW2[i]; }
  for (int i = tid; i < 64; i += 256)  { sB2[0][i] = kb2[i]; sB2[1][i] = fb2[i]; sW3[0][i] = kW3[i]; sW3[1][i] = fW3[i]; }
  if (tid == 0) { sB3[0] = kb3[0]; sB3[1] = fb3[0]; }
  __syncthreads();

  int idx = blockIdx.x * 256 + tid;
  int rr = idx >> 10;
  int cc = idx & (D_SZ - 1);
  const float step = 2.0f / 31.0f;
  float e[4];
  e[0] = -1.0f + step * (float)(rr >> 5);
  e[1] = -1.0f + step * (float)(rr & 31);
  e[2] = -1.0f + step * (float)(cc & 31);
  e[3] = -1.0f + step * (float)(cc >> 5);

  for (int m = 0; m < 2; m++) {
    float h1[32];
#pragma unroll
    for (int o = 0; o < 32; o++) {
      float s = sB1[m][o];
#pragma unroll
      for (int i2 = 0; i2 < 4; i2++) s += e[i2] * sW1[m][i2][o];
      h1[o] = s > 0.0f ? s : 0.01f * s;
    }
    float outv = sB3[m];
    for (int o = 0; o < 64; o++) {
      float s = sB2[m][o];
#pragma unroll
      for (int i2 = 0; i2 < 32; i2++) s += h1[i2] * sW2[m][i2][o];
      s = s > 0.0f ? s : 0.01f * s;
      outv += s * sW3[m][o];
    }
    (m ? wfT : wT)[(size_t)cc * D_SZ + rr] = f2bf(outv);
  }
}

// ---------------- Kernel 2: LayerNorm (per batch over 2048x64) + optional residual ----------------
__global__ void __launch_bounds__(256) ln_kernel(
    const float* src, const float* resid, int has_resid,
    const float* g, const float* bb,
    float* outf, __bf16* outb)
{
  __shared__ float rs[256], rs2[256];
  __shared__ float sm[2];
  int b = blockIdx.x, tid = threadIdx.x;
  const int N = TWO_D * IN_D;
  size_t base = (size_t)b * N;
  float s = 0.0f, s2 = 0.0f;
  for (int i = tid; i < N; i += 256) { float x = src[base + i]; s += x; s2 += x * x; }
  rs[tid] = s; rs2[tid] = s2;
  __syncthreads();
  for (int w = 128; w > 0; w >>= 1) {
    if (tid < w) { rs[tid] += rs[tid + w]; rs2[tid] += rs2[tid + w]; }
    __syncthreads();
  }
  if (tid == 0) {
    float mean = rs[0] / (float)N;
    float var = rs2[0] / (float)N - mean * mean;
    sm[0] = mean; sm[1] = rsqrtf(var + 1e-5f);
  }
  __syncthreads();
  float mean = sm[0], rstd = sm[1];
  for (int i = tid; i < N; i += 256) {
    float x = src[base + i];
    float v = (x - mean) * rstd * g[i] + bb[i];
    if (has_resid) v += resid[base + i];
    outf[base + i] = v;
    outb[base + i] = f2bf(v);
  }
}

// ---------------- Kernel 3: Q/K projection (WMMA): out = Vin @ W + bias -> bf16 ----------------
__global__ void __launch_bounds__(128) qk_project_kernel(
    const __bf16* Vin, const float* Wq, const float* bq,
    const float* Wk, const float* bk,
    __bf16* Qout, __bf16* Kout)
{
  __shared__ __align__(16) __bf16 lA[64 * 64];
  __shared__ __align__(16) __bf16 lB[64 * 64];
  int tid = threadIdx.x;
  int row0 = blockIdx.x * 64;
  int b = blockIdx.y;
  int head = blockIdx.z >> 1;
  int isK = blockIdx.z & 1;
  const float* W = (isK ? Wk : Wq) + (size_t)head * 64 * 64;
  const float* bias = (isK ? bk : bq) + (size_t)head * 64;
  __bf16* outp = (isK ? Kout : Qout) + ((size_t)(head * NBATCH + b)) * TWO_D * 64;

  const uint4* srcA = (const uint4*)(Vin + ((size_t)b * TWO_D + row0) * 64);
  uint4* dstA = (uint4*)lA;
  for (int t = tid; t < 512; t += 128) copy_b128(srcA + t, dstA + t);
  for (int i = tid; i < 4096; i += 128) {
    int k = i >> 6, n = i & 63;
    lB[n * 64 + k] = f2bf(W[i]);          // N-major staging of B
  }
  wait_async_copies();
  __syncthreads();

  int wv = tid >> 5, lane = tid & 31;
  v8f acc[4] = {};
#pragma unroll
  for (int ks = 0; ks < 2; ks++) {
    v16bf af = load_frag_a(lA + (wv * 16) * 64 + ks * 32, 64);
#pragma unroll
    for (int ct = 0; ct < 4; ct++) {
      v16bf bf = load_frag_b(lB + (ct * 16) * 64 + ks * 32, 64);
      acc[ct] = wmma_bf16(af, bf, acc[ct]);
    }
  }
  int n_c = lane & 15, mb = (lane >> 4) << 3;
#pragma unroll
  for (int ct = 0; ct < 4; ct++) {
    float bv = bias[ct * 16 + n_c];
#pragma unroll
    for (int v = 0; v < 8; v++)
      outp[((size_t)(row0 + wv * 16 + mb + v)) * 64 + ct * 16 + n_c] = f2bf(acc[ct][v] + bv);
  }
}

// ---------------- Kernel 4: M = K^T V (64x64 per head/batch, f32) ----------------
__global__ void __launch_bounds__(256) compute_m_kernel(
    const __bf16* Kbf, const float* Vsrc, int T, float* Mout)
{
  __shared__ float Kc[64][64];
  __shared__ float Vc[64][64];
  int head = blockIdx.x, b = blockIdx.y, tid = threadIdx.x;
  const __bf16* Kp = Kbf + ((size_t)(head * NBATCH + b)) * TWO_D * 64;
  const float* Vp = Vsrc + (size_t)b * TWO_D * 64;
  int g = tid & 63;
  int fb = (tid >> 6) * 16;
  float acc[16];
#pragma unroll
  for (int i = 0; i < 16; i++) acc[i] = 0.0f;
  for (int c0 = 0; c0 < T; c0 += 64) {
    for (int i = tid; i < 4096; i += 256) {
      int t = i >> 6, f = i & 63;
      Kc[t][f] = bf2f(Kp[((size_t)(c0 + t)) * 64 + f]);
      Vc[t][f] = Vp[((size_t)(c0 + t)) * 64 + f];
    }
    __syncthreads();
    for (int t = 0; t < 64; t++) {
      float vv = Vc[t][g];
#pragma unroll
      for (int ff = 0; ff < 16; ff++) acc[ff] += Kc[t][fb + ff] * vv;
    }
    __syncthreads();
  }
  float* Mp = Mout + ((size_t)(head * NBATCH + b)) * 4096;
  for (int ff = 0; ff < 16; ff++) Mp[(fb + ff) * 64 + g] = acc[ff];
}

// ---------------- Kernel 5: mid = cmid * sum_i Q_i @ M_i (WMMA) ----------------
__global__ void __launch_bounds__(128) apply_attn_kernel(
    const __bf16* Qbf, const float* Mm, float* mid, float cmid)
{
  __shared__ __align__(16) __bf16 lA[64 * 64];
  __shared__ __align__(16) __bf16 lB[64 * 64];
  int tid = threadIdx.x;
  int row0 = blockIdx.x * 64;
  int b = blockIdx.y;
  int wv = tid >> 5, lane = tid & 31;
  v8f acc[4] = {};
  for (int head = 0; head < NHEAD; head++) {
    const uint4* srcA = (const uint4*)(Qbf + (((size_t)(head * NBATCH + b)) * TWO_D + row0) * 64);
    uint4* dstA = (uint4*)lA;
    for (int t = tid; t < 512; t += 128) copy_b128(srcA + t, dstA + t);
    const float* Mp = Mm + ((size_t)(head * NBATCH + b)) * 4096;
    for (int i = tid; i < 4096; i += 128) {
      int k = i >> 6, n = i & 63;
      lB[n * 64 + k] = f2bf(Mp[i]);
    }
    wait_async_copies();
    __syncthreads();
#pragma unroll
    for (int ks = 0; ks < 2; ks++) {
      v16bf af = load_frag_a(lA + (wv * 16) * 64 + ks * 32, 64);
#pragma unroll
      for (int ct = 0; ct < 4; ct++) {
        v16bf bf = load_frag_b(lB + (ct * 16) * 64 + ks * 32, 64);
        acc[ct] = wmma_bf16(af, bf, acc[ct]);
      }
    }
    __syncthreads();
  }
  int n_c = lane & 15, mb = (lane >> 4) << 3;
  for (int ct = 0; ct < 4; ct++)
    for (int v = 0; v < 8; v++)
      mid[((size_t)b * TWO_D + row0 + wv * 16 + mb + v) * 64 + ct * 16 + n_c] = cmid * acc[ct][v];
}

// ---------------- Kernel 6: Tu/Tf = sum_i Q[qoff..] @ M_i, stored TRANSPOSED [c][token] ----------------
__global__ void __launch_bounds__(128) accum_t_kernel(
    const __bf16* Qbf, const float* Mm, __bf16* Tu, __bf16* Tf)
{
  __shared__ __align__(16) __bf16 lA[64 * 64];
  __shared__ __align__(16) __bf16 lB[64 * 64];
  int tid = threadIdx.x;
  int row0 = blockIdx.x * 64;           // token within D
  int b = blockIdx.y;
  int qoff = blockIdx.z * D_SZ;         // 0 = upper (u), D = lower (f)
  int wv = tid >> 5, lane = tid & 31;
  v8f acc[4] = {};
  for (int head = 0; head < NHEAD; head++) {
    const uint4* srcA = (const uint4*)(Qbf + (((size_t)(head * NBATCH + b)) * TWO_D + qoff + row0) * 64);
    uint4* dstA = (uint4*)lA;
    for (int t = tid; t < 512; t += 128) copy_b128(srcA + t, dstA + t);
    const float* Mp = Mm + ((size_t)(head * NBATCH + b)) * 4096;
    for (int i = tid; i < 4096; i += 128) {
      int k = i >> 6, n = i & 63;
      lB[n * 64 + k] = f2bf(Mp[i]);
    }
    wait_async_copies();
    __syncthreads();
#pragma unroll
    for (int ks = 0; ks < 2; ks++) {
      v16bf af = load_frag_a(lA + (wv * 16) * 64 + ks * 32, 64);
#pragma unroll
      for (int ct = 0; ct < 4; ct++) {
        v16bf bf = load_frag_b(lB + (ct * 16) * 64 + ks * 32, 64);
        acc[ct] = wmma_bf16(af, bf, acc[ct]);
      }
    }
    __syncthreads();
  }
  // Transposed store: T[b][c][token] so the final GEMM can stage B contiguously.
  __bf16* outp = (blockIdx.z ? Tf : Tu) + (size_t)b * D_SZ * 64;
  int n_c = lane & 15, mb = (lane >> 4) << 3;
  for (int ct = 0; ct < 4; ct++)
    for (int v = 0; v < 8; v++)
      outp[(size_t)(ct * 16 + n_c) * D_SZ + (row0 + wv * 16 + mb + v)] = f2bf(acc[ct][v]);
}

// ---------------- Kernel 7: out = cf * (wT @ Tu + wfT @ Tf) (WMMA, K=1024) ----------------
__global__ void __launch_bounds__(128) final_gemm_kernel(
    const __bf16* wT, const __bf16* wfT,
    const __bf16* Tu, const __bf16* Tf,
    float* outp, float cf)
{
  __shared__ __align__(16) __bf16 lA[64 * 32];
  __shared__ __align__(16) __bf16 lB[64 * 32];
  int tid = threadIdx.x;
  int n0 = blockIdx.x * 64;
  int b = blockIdx.y;
  int wv = tid >> 5, lane = tid & 31;
  v8f acc[4] = {};
  for (int w2 = 0; w2 < 2; w2++) {
    const __bf16* Ap = w2 ? wfT : wT;
    const __bf16* Bp = (w2 ? Tf : Tu) + (size_t)b * D_SZ * 64;   // [c][token]
    for (int k = 0; k < D_SZ; k += 32) {
      // stage A: rows n0..n0+63, K-slice k..k+31 (contiguous 16B chunks)
      for (int ci = tid; ci < 256; ci += 128) {
        int rrow = ci >> 2, cp = ci & 3;
        copy_b128(Ap + (size_t)(n0 + rrow) * D_SZ + k + cp * 8,
                  lA + rrow * 32 + cp * 8);
      }
      // stage B (already N-major in memory): contiguous 16B chunks too
      for (int ci = tid; ci < 256; ci += 128) {
        int c = ci >> 2, cp = ci & 3;
        copy_b128(Bp + (size_t)c * D_SZ + k + cp * 8,
                  lB + c * 32 + cp * 8);
      }
      wait_async_copies();
      __syncthreads();
      v16bf af = load_frag_a(lA + (wv * 16) * 32, 32);
#pragma unroll
      for (int ct = 0; ct < 4; ct++) {
        v16bf bf = load_frag_b(lB + (ct * 16) * 32, 32);
        acc[ct] = wmma_bf16(af, bf, acc[ct]);
      }
      __syncthreads();
      if (k + 32 < D_SZ)
        __builtin_prefetch((const void*)(Ap + (size_t)(n0 + (tid >> 2)) * D_SZ + k + 32), 0, 1);
    }
  }
  int n_c = lane & 15, mb = (lane >> 4) << 3;
  for (int ct = 0; ct < 4; ct++)
    for (int v = 0; v < 8; v++)
      outp[((size_t)b * D_SZ + n0 + wv * 16 + mb + v) * 64 + ct * 16 + n_c] = cf * acc[ct][v];
}

extern "C" void kernel_launch(void* const* d_in, const int* in_sizes, int n_in,
                              void* d_out, int out_size, void* d_ws, size_t ws_size,
                              hipStream_t stream)
{
  (void)in_sizes; (void)n_in; (void)out_size; (void)ws_size;
  const float* xy   = (const float*)d_in[1];
  const float* kW1  = (const float*)d_in[3];
  const float* kb1  = (const float*)d_in[4];
  const float* kW2  = (const float*)d_in[5];
  const float* kb2  = (const float*)d_in[6];
  const float* kW3  = (const float*)d_in[7];
  const float* kb3  = (const float*)d_in[8];
  const float* fW1  = (const float*)d_in[9];
  const float* fb1  = (const float*)d_in[10];
  const float* fW2  = (const float*)d_in[11];
  const float* fb2  = (const float*)d_in[12];
  const float* fW3  = (const float*)d_in[13];
  const float* fb3  = (const float*)d_in[14];
  const float* Wq   = (const float*)d_in[15];
  const float* bq   = (const float*)d_in[16];
  const float* Wk   = (const float*)d_in[17];
  const float* bk   = (const float*)d_in[18];
  const float* ln_g = (const float*)d_in[19];
  const float* ln_b = (const float*)d_in[20];

  char* ws = (char*)d_ws;
  size_t off = 0;
  auto alloc = [&](size_t bytes) -> void* {
    void* p = ws + off;
    off += (bytes + 255) & ~(size_t)255;
    return p;
  };
  __bf16* wT   = (__bf16*)alloc((size_t)D_SZ * D_SZ * 2);
  __bf16* wfT  = (__bf16*)alloc((size_t)D_SZ * D_SZ * 2);
  float*  Vinf = (float*) alloc((size_t)NBATCH * TWO_D * IN_D * 4);
  __bf16* Vinb = (__bf16*)alloc((size_t)NBATCH * TWO_D * IN_D * 2);
  float*  mid  = (float*) alloc((size_t)NBATCH * TWO_D * IN_D * 4);
  __bf16* Qb   = (__bf16*)alloc((size_t)NHEAD * NBATCH * TWO_D * IN_D * 2);
  __bf16* Kb   = (__bf16*)alloc((size_t)NHEAD * NBATCH * TWO_D * IN_D * 2);
  float*  Mm   = (float*) alloc((size_t)NHEAD * NBATCH * 64 * 64 * 4);
  __bf16* Tu   = (__bf16*)alloc((size_t)NBATCH * D_SZ * IN_D * 2);
  __bf16* Tf   = (__bf16*)alloc((size_t)NBATCH * D_SZ * IN_D * 2);

  const float DXv   = 2.0f / 31.0f;
  const float DXDY  = DXv * DXv;
  const float SCALE = 0.125f;             // 1/sqrt(64)
  const float CMID  = DXDY * SCALE;
  const float CF    = DXDY * DXDY * SCALE;

  mlp_edge_kernel<<<4096, 256, 0, stream>>>(kW1, kb1, kW2, kb2, kW3, kb3,
                                            fW1, fb1, fW2, fb2, fW3, fb3, wT, wfT);
  ln_kernel<<<8, 256, 0, stream>>>(xy, nullptr, 0, ln_g, ln_b, Vinf, Vinb);

  for (int j = 0; j < 2; j++) {
    qk_project_kernel<<<dim3(32, 8, 8), 128, 0, stream>>>(
        Vinb, Wq + (size_t)j * NHEAD * 64 * 64, bq + (size_t)j * NHEAD * 64,
        Wk + (size_t)j * NHEAD * 64 * 64, bk + (size_t)j * NHEAD * 64, Qb, Kb);
    compute_m_kernel<<<dim3(NHEAD, 8), 256, 0, stream>>>(Kb, Vinf, TWO_D, Mm);
    apply_attn_kernel<<<dim3(32, 8), 128, 0, stream>>>(Qb, Mm, mid, CMID);
    ln_kernel<<<8, 256, 0, stream>>>(mid, Vinf, 1,
                                     ln_g + (size_t)(j + 1) * TWO_D * IN_D,
                                     ln_b + (size_t)(j + 1) * TWO_D * IN_D, Vinf, Vinb);
  }

  const int j2 = 2;
  qk_project_kernel<<<dim3(32, 8, 8), 128, 0, stream>>>(
      Vinb, Wq + (size_t)j2 * NHEAD * 64 * 64, bq + (size_t)j2 * NHEAD * 64,
      Wk + (size_t)j2 * NHEAD * 64 * 64, bk + (size_t)j2 * NHEAD * 64, Qb, Kb);
  compute_m_kernel<<<dim3(NHEAD, 8), 256, 0, stream>>>(Kb, xy, D_SZ, Mm);
  accum_t_kernel<<<dim3(16, 8, 2), 128, 0, stream>>>(Qb, Mm, Tu, Tf);
  final_gemm_kernel<<<dim3(16, 8), 128, 0, stream>>>(wT, wfT, Tu, Tf, (float*)d_out, CF);
}
